// DotProductAttention_84181359001856
// MI455X (gfx1250) — matrix operand order repaired and checked by reading, not compiled, not run on hardware
//
#include <hip/hip_runtime.h>
#include <hip/hip_bf16.h>

typedef __attribute__((ext_vector_type(16))) _Float16     v16h;
typedef __attribute__((ext_vector_type(8)))  float        v8f;
typedef __attribute__((ext_vector_type(4)))  unsigned int v4u;
typedef __attribute__((ext_vector_type(8)))  int          v8i;
typedef __attribute__((ext_vector_type(4)))  int          v4i;

#define S_LEN 2048
#define D_DIM 64
#define WAVES 8
#define QT    16    // query rows per wave
#define KBB   64    // keys per block
#define NELEM (4 * 16 * S_LEN * D_DIM)   // 8,388,608 per tensor

union AV { v16h h; unsigned int u[8]; };

__device__ __forceinline__ unsigned int pack2(float a, float b) {
    union { _Float16 h[2]; unsigned int u; } x;
    x.h[0] = (_Float16)a; x.h[1] = (_Float16)b;
    return x.u;
}

__device__ __forceinline__ v8f wmma16(v16h a, v16h b, v8f c) {
    return __builtin_amdgcn_wmma_f32_16x16x32_f16(
        false, a, false, b, (short)0, c, false, false);
}

__device__ __forceinline__ float rmax16(float v) {
    #pragma unroll
    for (int m = 8; m >= 1; m >>= 1) v = fmaxf(v, __shfl_xor(v, m, 32));
    return v;
}
__device__ __forceinline__ float rsum16(float v) {
    #pragma unroll
    for (int m = 8; m >= 1; m >>= 1) v += __shfl_xor(v, m, 32);
    return v;
}

// ---- TDM: 2D f16 tile, global -> LDS. D# built per CDNA5 ISA 08_async_tensor §8 ----
__device__ __forceinline__ void tdm_load_2d_f16(unsigned lds_addr, const void* gptr,
                                                unsigned tensor_d0, unsigned tensor_d1,
                                                unsigned tile_d0, unsigned tile_d1,
                                                unsigned long long stride0_elems) {
    unsigned long long ga = (unsigned long long)(uintptr_t)gptr;
    v4u g0;
    g0[0] = 1u;                                        // count=1, user D#
    g0[1] = lds_addr;                                  // bits 63:32
    g0[2] = (unsigned)(ga & 0xffffffffu);              // global_addr[31:0]
    g0[3] = (unsigned)((ga >> 32) & 0x1ffffffu)        // global_addr[56:32]
          | (2u << 30);                                // type=2 ("image")
    v8i g1;
    g1[0] = (int)(1u << 16);                           // data_size=1 -> 2 bytes; wg_mask=0
    g1[1] = (int)((tensor_d0 & 0xffffu) << 16);        // tensor_dim0[15:0] @ 63:48
    g1[2] = (int)(((tensor_d0 >> 16) & 0xffffu) | ((tensor_d1 & 0xffffu) << 16));
    g1[3] = (int)(((tensor_d1 >> 16) & 0xffffu) | ((tile_d0 & 0xffffu) << 16));
    g1[4] = (int)(tile_d1 & 0xffffu);                  // tile_dim1; tile_dim2=0
    g1[5] = (int)(unsigned)(stride0_elems & 0xffffffffu);       // dim0 stride lo
    g1[6] = (int)(unsigned)((stride0_elems >> 32) & 0xffffu);   // dim0 stride hi
    g1[7] = 0;
    v4i z4 = {0, 0, 0, 0};
    v8i z8 = {0, 0, 0, 0, 0, 0, 0, 0};
    // amdgpu-toolchain (clang-23) 6-arg form: (g0, g1, g2, g3, extra, cpol)
    __builtin_amdgcn_tensor_load_to_lds(g0, g1, z4, z4, z8, 0);
}

// ---- pre-pass: fp32 -> f16 straight copy (for K) ----
__global__ __launch_bounds__(256)
void cvt_f16_kernel(const float* __restrict__ in, _Float16* __restrict__ out) {
    const size_t i = (size_t)(blockIdx.x * 256 + threadIdx.x) * 8;
    float4 a = *(const float4*)(in + i);
    float4 b = *(const float4*)(in + i + 4);
    uint4 u;
    u.x = pack2(a.x, a.y); u.y = pack2(a.z, a.w);
    u.z = pack2(b.x, b.y); u.w = pack2(b.z, b.w);
    *(uint4*)(out + i) = u;
}

// ---- pre-pass: V fp32 [bh][s][d] -> f16 transposed [bh][d][s] ----
__global__ __launch_bounds__(256)
void v_transpose_f16(const float* __restrict__ V, _Float16* __restrict__ Vt) {
    const int t  = blockIdx.x * 256 + threadIdx.x;   // 64*2048*16 threads
    const int d4 = t & 15;
    const int s  = (t >> 4) & (S_LEN - 1);
    const int bh = t >> 15;
    float4 v4 = *(const float4*)(V + ((size_t)bh * S_LEN + s) * D_DIM + d4 * 4);
    const size_t o = ((size_t)bh * D_DIM + d4 * 4) * S_LEN + s;
    Vt[o]             = (_Float16)v4.x;
    Vt[o + S_LEN]     = (_Float16)v4.y;
    Vt[o + 2 * S_LEN] = (_Float16)v4.z;
    Vt[o + 3 * S_LEN] = (_Float16)v4.w;
}

__global__ __launch_bounds__(256)
void fa_causal_wmma(const float* __restrict__ Q,
                    const _Float16* __restrict__ Kh,   // [bh][s][d] f16
                    const _Float16* __restrict__ Vt,   // [bh][d][s] f16
                    float* __restrict__ O) {
    // double-buffered TDM tiles
    __shared__ _Float16 sK[2][KBB * D_DIM];         // 2 x 8 KB  [key][d]
    __shared__ _Float16 sV[2][D_DIM * KBB];         // 2 x 8 KB  [d][key]
    __shared__ _Float16 sQ[WAVES * QT * D_DIM];     // 16 KB per-wave [m][d]
    __shared__ _Float16 sP[WAVES * QT * KBB];       // 16 KB per-wave [m][k]

    const int tid  = threadIdx.x;
    const int wave = tid >> 5;
    const int lane = tid & 31;
    const int ln   = lane & 15;
    const int hi   = lane >> 4;

    const int bh = blockIdx.y;
    const int qb = blockIdx.x * (WAVES * QT) + wave * QT;

    const float*    Qb  = Q  + (size_t)bh * S_LEN * D_DIM;
    const _Float16* Kbh = Kh + (size_t)bh * S_LEN * D_DIM;
    const _Float16* Vbh = Vt + (size_t)bh * D_DIM * S_LEN;
    float*          Ob  = O  + (size_t)bh * S_LEN * D_DIM;

    _Float16* sQw = sQ + wave * QT * D_DIM;
    _Float16* sPw = sP + wave * QT * KBB;

    // ---- stage this wave's Q tile (16x64 fp32 -> f16, row-major) ----
    #pragma unroll
    for (int j = 0; j < 8; ++j) {
        const int e = lane * 4 + j * 128;
        float4 q4 = *(const float4*)(Qb + (size_t)qb * D_DIM + e);
        unsigned long long u = ((unsigned long long)pack2(q4.z, q4.w) << 32)
                             |  (unsigned long long)pack2(q4.x, q4.y);
        *(unsigned long long*)(&sQw[e]) = u;
    }
    asm volatile("s_wait_dscnt 0" ::: "memory");

    // ---- Q in WMMA A-layout: half t=2v+h -> k = t + (t>=8?8:0) + 8*hi ----
    AV qa0, qa1;
    #pragma unroll
    for (int v = 0; v < 8; ++v) {
        const int t = 2 * v;
        const int k = t + ((t >= 8) ? 8 : 0) + 8 * hi;
        qa0.u[v] = *(const unsigned int*)(&sQw[ln * D_DIM + k]);
        qa1.u[v] = *(const unsigned int*)(&sQw[ln * D_DIM + 32 + k]);
    }

    float rm[8], rl[8];
    #pragma unroll
    for (int i = 0; i < 8; ++i) { rm[i] = -1e30f; rl[i] = 0.0f; }
    v8f acc0 = {}, acc1 = {}, acc2 = {}, acc3 = {};

    const float kScale = 0.18033688011112042f;          // log2(e)/sqrt(64)
    const unsigned ldsK[2] = { (unsigned)(uintptr_t)&sK[0][0],
                               (unsigned)(uintptr_t)&sK[1][0] };
    const unsigned ldsV[2] = { (unsigned)(uintptr_t)&sV[0][0],
                               (unsigned)(uintptr_t)&sV[1][0] };
    const int nkb = (blockIdx.x + 1) * ((WAVES * QT) / KBB);

    // ---- prologue: kick off DMA of block 0 into buffer 0 ----
    if (wave == 0) {
        tdm_load_2d_f16(ldsK[0], Kbh, D_DIM, KBB, D_DIM, KBB, D_DIM);
        tdm_load_2d_f16(ldsV[0], Vbh, KBB, D_DIM, KBB, D_DIM, S_LEN);
    }

    for (int kb = 0; kb < nkb; ++kb) {
        const int cur = kb & 1;
        // current buffer's DMA complete; barrier also certifies that the
        // alternate buffer's readers (iteration kb-1) are finished.
        if (wave == 0) __builtin_amdgcn_s_wait_tensorcnt(0);
        __syncthreads();
        // overlap: DMA next block into the alternate buffer while computing
        if (wave == 0 && (kb + 1) < nkb) {
            const int nxt = 1 - cur;
            tdm_load_2d_f16(ldsK[nxt], Kbh + (size_t)((kb + 1) * KBB) * D_DIM,
                            D_DIM, KBB, D_DIM, KBB, D_DIM);
            tdm_load_2d_f16(ldsV[nxt], Vbh + (size_t)((kb + 1) * KBB),
                            KBB, D_DIM, KBB, D_DIM, S_LEN);
        }

        if (kb * KBB > qb + (QT - 1)) continue;   // fully masked for this wave

        const _Float16* cK = sK[cur];
        const _Float16* cV = sV[cur];

        // ---- scores: 4 key-tiles of 16, each accumulated over d in 2 chunks ----
        v8f sc0 = {}, sc1 = {}, sc2 = {}, sc3 = {};
        AV b;
        #pragma unroll
        for (int kt = 0; kt < 4; ++kt) {
            #pragma unroll
            for (int v = 0; v < 8; ++v)
                b.u[v] = *(const unsigned int*)(&cK[(kt * 16 + ln) * D_DIM + (2 * v + ((v >= 4) ? 8 : 0) + 8 * hi)]);
            v8f* sc = (kt == 0) ? &sc0 : (kt == 1) ? &sc1 : (kt == 2) ? &sc2 : &sc3;
            *sc = wmma16(qa0.h, b.h, *sc);
            #pragma unroll
            for (int v = 0; v < 8; ++v)
                b.u[v] = *(const unsigned int*)(&cK[(kt * 16 + ln) * D_DIM + 32 + (2 * v + ((v >= 4) ? 8 : 0) + 8 * hi)]);
            *sc = wmma16(qa1.h, b.h, *sc);
        }

        // ---- online softmax (base-2) with causal mask on diagonal blocks ----
        const bool needMask = (kb * KBB + KBB - 1) > qb;
        const int k0 = kb * KBB + ln;
        #pragma unroll
        for (int i = 0; i < 8; ++i) {
            const int qg = qb + i + 8 * hi;
            float a0 = sc0[i] * kScale;
            float a1 = sc1[i] * kScale;
            float a2 = sc2[i] * kScale;
            float a3 = sc3[i] * kScale;
            if (needMask) {
                if (k0      > qg) a0 = -1e30f;
                if (k0 + 16 > qg) a1 = -1e30f;
                if (k0 + 32 > qg) a2 = -1e30f;
                if (k0 + 48 > qg) a3 = -1e30f;
            }
            const float mx   = rmax16(fmaxf(fmaxf(a0, a1), fmaxf(a2, a3)));
            const float mnew = fmaxf(rm[i], mx);
            const float scl  = exp2f(rm[i] - mnew);
            const float p0   = exp2f(a0 - mnew);
            const float p1   = exp2f(a1 - mnew);
            const float p2   = exp2f(a2 - mnew);
            const float p3   = exp2f(a3 - mnew);
            rl[i] = rl[i] * scl + rsum16((p0 + p1) + (p2 + p3));
            rm[i] = mnew;
            acc0[i] *= scl; acc1[i] *= scl; acc2[i] *= scl; acc3[i] *= scl;
            const int row = i + 8 * hi;
            sPw[row * KBB + ln]      = (_Float16)p0;
            sPw[row * KBB + 16 + ln] = (_Float16)p1;
            sPw[row * KBB + 32 + ln] = (_Float16)p2;
            sPw[row * KBB + 48 + ln] = (_Float16)p3;
        }
        asm volatile("s_wait_dscnt 0" ::: "memory");

        // ---- P (16x64) as two A-matrices, then O += P * V ----
        AV pa0, pa1;
        #pragma unroll
        for (int v = 0; v < 8; ++v) {
            const int t = 2 * v;
            const int k = t + ((t >= 8) ? 8 : 0) + 8 * hi;
            pa0.u[v] = *(const unsigned int*)(&sPw[ln * KBB + k]);
            pa1.u[v] = *(const unsigned int*)(&sPw[ln * KBB + 32 + k]);
        }
        AV vb;
        #pragma unroll
        for (int dt = 0; dt < 4; ++dt) {
            v8f* ac = (dt == 0) ? &acc0 : (dt == 1) ? &acc1 : (dt == 2) ? &acc2 : &acc3;
            #pragma unroll
            for (int v = 0; v < 8; ++v)
                vb.u[v] = *(const unsigned int*)(&cV[(dt * 16 + ln) * KBB + (2 * v + ((v >= 4) ? 8 : 0) + 8 * hi)]);
            *ac = wmma16(pa0.h, vb.h, *ac);
            #pragma unroll
            for (int v = 0; v < 8; ++v)
                vb.u[v] = *(const unsigned int*)(&cV[(dt * 16 + ln) * KBB + 32 + (2 * v + ((v >= 4) ? 8 : 0) + 8 * hi)]);
            *ac = wmma16(pa1.h, vb.h, *ac);
        }
    }

    // ---- normalize and store fp32 output ----
    #pragma unroll
    for (int i = 0; i < 8; ++i) {
        const float inv = 1.0f / rl[i];
        const size_t o = (size_t)(qb + i + 8 * hi) * D_DIM + ln;
        Ob[o]      = acc0[i] * inv;
        Ob[o + 16] = acc1[i] * inv;
        Ob[o + 32] = acc2[i] * inv;
        Ob[o + 48] = acc3[i] * inv;
    }
}

extern "C" void kernel_launch(void* const* d_in, const int* in_sizes, int n_in,
                              void* d_out, int out_size, void* d_ws, size_t ws_size,
                              hipStream_t stream) {
    const float* Q = (const float*)d_in[0];
    const float* K = (const float*)d_in[1];
    const float* V = (const float*)d_in[2];
    float* O = (float*)d_out;

    // workspace: f16 K (row-major) + f16 V (transposed) = 2 * 16 MB
    _Float16* Kh = (_Float16*)d_ws;
    _Float16* Vt = Kh + (size_t)NELEM;

    cvt_f16_kernel<<<NELEM / (256 * 8), 256, 0, stream>>>(K, Kh);
    v_transpose_f16<<<(64 * S_LEN * 16) / 256, 256, 0, stream>>>(V, Vt);

    const int B = 4, H = 16;
    dim3 grid(S_LEN / (WAVES * QT), B * H);   // (16, 64)
    dim3 block(WAVES * 32);                   // 8 wave32
    fa_causal_wmma<<<grid, block, 0, stream>>>(Q, Kh, Vt, O);
}
